// SwitchMoE_75883482185740
// MI455X (gfx1250) — matrix operand order repaired and checked by reading, not compiled, not run on hardware
//
#include <hip/hip_runtime.h>
#include <math.h>

#define NC 768      // C
#define NH 3072     // H
#define NE 8        // experts
#define NT 8192     // tokens (B*T)
#define CHUNK 128   // H chunk per iteration
#define MT 32       // tokens per block tile
#define LDX (NC + 8)     // LDS row stride (bf16 elems) for x tile
#define LDH (CHUNK + 8)  // LDS row stride (bf16 elems) for h tile

typedef __attribute__((ext_vector_type(16))) __bf16 v16bf;
typedef __attribute__((ext_vector_type(8)))  float  v8f;

union FragA { v16bf v; uint4 q[2]; };
union FragB { v16bf v; uint4 q[2]; __bf16 b[16]; };

__device__ __forceinline__ void loadBfrag(FragB& B, const __bf16* base) {
    const uint4* p = (const uint4*)base;
    B.q[0] = p[0]; B.q[1] = p[1];
}
__device__ __forceinline__ void loadAfrag(FragA& A, const __bf16* base) {
    A.q[0] = *(const uint4*)base;
    A.q[1] = *(const uint4*)(base + 8);
}

// ---------------- init: zero expert counters ----------------
__global__ void moe_init_kernel(int* cnts) {
    if (threadIdx.x < NE) cnts[threadIdx.x] = 0;
}

// ---------------- router: logits, softmax, top-1, lists ----------------
__global__ void moe_router_kernel(const float* __restrict__ x,
                                  const float* __restrict__ Wr,
                                  int* __restrict__ cnts,
                                  int* __restrict__ elist,
                                  float* __restrict__ tprob,
                                  float* __restrict__ impPart) {
    const int wv = threadIdx.x >> 5;
    const int lane = threadIdx.x & 31;
    const int tok = blockIdx.x * 8 + wv;

    float acc[NE];
#pragma unroll
    for (int e = 0; e < NE; ++e) acc[e] = 0.0f;

    const float* xr = x + (size_t)tok * NC;
    for (int c = lane; c < NC; c += 32) {
        float xv = xr[c];
        const float4* w = (const float4*)(Wr + (size_t)c * NE);
        float4 w0 = w[0], w1 = w[1];
        acc[0] += xv * w0.x; acc[1] += xv * w0.y;
        acc[2] += xv * w0.z; acc[3] += xv * w0.w;
        acc[4] += xv * w1.x; acc[5] += xv * w1.y;
        acc[6] += xv * w1.z; acc[7] += xv * w1.w;
    }
#pragma unroll
    for (int e = 0; e < NE; ++e)
#pragma unroll
        for (int off = 16; off > 0; off >>= 1)
            acc[e] += __shfl_xor(acc[e], off, 32);

    __shared__ float wp[8][NE];
    if (lane == 0) {
        float mx = acc[0];
#pragma unroll
        for (int e = 1; e < NE; ++e) mx = fmaxf(mx, acc[e]);
        float p[NE], s = 0.0f;
#pragma unroll
        for (int e = 0; e < NE; ++e) { p[e] = expf(acc[e] - mx); s += p[e]; }
        float inv = 1.0f / s;
        int best = 0; float bl = acc[0];
#pragma unroll
        for (int e = 1; e < NE; ++e) if (acc[e] > bl) { bl = acc[e]; best = e; }
        int pos = atomicAdd(&cnts[best], 1);
        elist[best * NT + pos] = tok;
        tprob[tok] = p[best] * inv;
#pragma unroll
        for (int e = 0; e < NE; ++e) wp[wv][e] = p[e] * inv;
    }
    __syncthreads();
    if (threadIdx.x < NE) {          // fixed-order per-block importance sum
        float s = 0.0f;
#pragma unroll
        for (int w = 0; w < 8; ++w) s += wp[w][threadIdx.x];
        impPart[blockIdx.x * NE + threadIdx.x] = s;
    }
}

// ---------------- aux loss: deterministic reduction ----------------
__global__ void moe_aux_kernel(const float* __restrict__ impPart,
                               const int* __restrict__ cnts,
                               float* __restrict__ out_aux) {
    __shared__ float imp[NE];
    if (threadIdx.x < NE) {
        float s = 0.0f;
        for (int b = 0; b < NT / 8; ++b) s += impPart[b * NE + threadIdx.x];
        imp[threadIdx.x] = s;
    }
    __syncthreads();
    if (threadIdx.x == 0) {
        float a = 0.0f;
        for (int e = 0; e < NE; ++e) a += imp[e] * (float)cnts[e];
        out_aux[0] = (float)NE * a / ((float)NT * (float)NT);
    }
}

// ---------------- weight pre-pack: fp32 -> bf16 in B-fragment order ----------
// Fragment layout: [e][ktile][ntile][lane(32)][16 bf16 along K]
// lane&15 = column within 16-wide N tile; lane>>4 selects K half (16).
__global__ void moe_pack_kernel(const float* __restrict__ W,
                                __bf16* __restrict__ out,
                                int ktiles, int ntiles, int ld) {
    size_t t = (size_t)blockIdx.x * 256 + threadIdx.x; // one lane of one fragment
    int lane = (int)(t & 31);
    size_t frag = t >> 5;
    int nt = (int)(frag % ntiles); size_t tmp = frag / ntiles;
    int kt = (int)(tmp % ktiles); int e = (int)(tmp / ktiles);
    int n  = nt * 16 + (lane & 15);
    int k0 = kt * 32 + (lane >> 4) * 16;
    const float* src = W + ((size_t)e * (ktiles * 32) + k0) * ld + n;
    __bf16* d = out + t * 16;
#pragma unroll
    for (int i = 0; i < 16; ++i) d[i] = (__bf16)src[(size_t)i * ld];
}

// ---------------- expert FFN: bf16 WMMA, 32-token tiles ----------------
// Each B-fragment (weights) feeds 2 WMMAs (two 16-row A tiles); B-fragments
// are double-buffered so the L2 load of B(i+1) overlaps the WMMAs on B(i).
template <bool PACKED>
__global__ void __launch_bounds__(256)
moe_ffn_kernel(const float* __restrict__ x,
               const float* __restrict__ W1, const float* __restrict__ b1,
               const float* __restrict__ W2, const float* __restrict__ b2,
               const __bf16* __restrict__ w1p, const __bf16* __restrict__ w2p,
               const int* __restrict__ cnts, const int* __restrict__ elist,
               const float* __restrict__ tprob, float* __restrict__ out) {
    const int e    = blockIdx.x >> 8;
    const int tile = blockIdx.x & 255;
    const int cnt  = cnts[e];
    const int m0   = tile * MT;
    if (m0 >= cnt) return;
    const int nTok = min(MT, cnt - m0);

    __shared__ __align__(16) __bf16 sx[MT * LDX];   // 49,664 B
    __shared__ __align__(16) __bf16 sh[MT * LDH];   //  8,704 B
    __shared__ int   stok[MT];
    __shared__ float sprob[MT];

    const int tid = threadIdx.x;
    if (tid < MT) {
        int tk = elist[e * NT + m0 + (tid < nTok ? tid : 0)];
        stok[tid]  = tk;
        sprob[tid] = tprob[tk];
    }
    __syncthreads();

    // stage x tile -> LDS bf16 (coalesced float4 reads)
    for (int idx = tid; idx < MT * (NC / 4); idx += 256) {
        int r = idx / (NC / 4), q = idx % (NC / 4);
        float4 xv = *(((const float4*)(x + (size_t)stok[r] * NC)) + q);
        __bf16* d = &sx[r * LDX + q * 4];
        d[0] = (__bf16)xv.x; d[1] = (__bf16)xv.y;
        d[2] = (__bf16)xv.z; d[3] = (__bf16)xv.w;
    }
    __syncthreads();

    const int wv  = tid >> 5;
    const int ln  = tid & 31;
    const int lh  = ln & 15;
    const int kup = ln >> 4;

    const float* W1e = W1 + (size_t)e * NC * NH;
    const float* W2e = W2 + (size_t)e * NH * NC;
    const float* b1e = b1 + (size_t)e * NH;
    const float* b2e = b2 + (size_t)e * NC;

    const v8f vzero = {0.f, 0.f, 0.f, 0.f, 0.f, 0.f, 0.f, 0.f};
    v8f yacc[2][6];
#pragma unroll
    for (int r = 0; r < 2; ++r)
#pragma unroll
        for (int t = 0; t < 6; ++t) yacc[r][t] = vzero;

    for (int ch = 0; ch < NH / CHUNK; ++ch) {
        // ---- GEMM1: h[32 x 16] for this wave's column tile ----
        const int nt1 = ch * (CHUNK / 16) + wv;      // N-tile index in H
        const int n1  = nt1 * 16 + lh;
        // B base for this wave's column, stepped by ks
        const __bf16* b1base = w1p +
            (((size_t)e * (NC / 32) * (NH / 16) + nt1) * 32 + ln) * 16;
        const size_t b1step = (size_t)(NH / 16) * 512;   // elems between ks tiles

        v8f hacc[2] = {vzero, vzero};
        FragB Bd[2];
        if (PACKED) loadBfrag(Bd[0], b1base);
        else {
            const float* bp = W1e + (size_t)(kup * 16) * NH + n1;
#pragma unroll
            for (int k = 0; k < 16; ++k) Bd[0].b[k] = (__bf16)bp[(size_t)k * NH];
        }
#pragma unroll
        for (int ks = 0; ks < NC / 32; ++ks) {
            const int cur = ks & 1;
            if (ks + 1 < NC / 32) {                  // prefetch next B
                if (PACKED) loadBfrag(Bd[cur ^ 1], b1base + (size_t)(ks + 1) * b1step);
                else {
                    const float* bp = W1e + (size_t)((ks + 1) * 32 + kup * 16) * NH + n1;
#pragma unroll
                    for (int k = 0; k < 16; ++k) Bd[cur ^ 1].b[k] = (__bf16)bp[(size_t)k * NH];
                }
            }
#pragma unroll
            for (int r = 0; r < 2; ++r) {
                FragA A;
                loadAfrag(A, &sx[(r * 16 + lh) * LDX + ks * 32 + kup * 16]);
                hacc[r] = __builtin_amdgcn_wmma_f32_16x16x32_bf16(
                    false, A.v, false, Bd[cur].v, (short)0, hacc[r], false, false);
            }
        }
        const float bias1 = b1e[n1];
#pragma unroll
        for (int r = 0; r < 2; ++r)
#pragma unroll
            for (int v = 0; v < 8; ++v) {
                float hv = hacc[r][v] + bias1;
                hv = 0.5f * hv * (1.0f + erff(hv * 0.70710678118654752f)); // exact GELU
                sh[(r * 16 + v + kup * 8) * LDH + wv * 16 + lh] = (__bf16)hv;
            }
        __syncthreads();

        // ---- GEMM2: accumulate y over this H chunk (flattened ks x t loop,
        //      B double-buffered across the whole 24-iteration space) ----
        const __bf16* b2base = w2p +
            ((((size_t)e * (NH / 32) + ch * (CHUNK / 32)) * (NC / 16) + wv * 6) * 32 + ln) * 16;
        // elems: step t -> 512, step ks -> (NC/16)*512 with t reset
        FragB Be[2];
        FragA A2[2];
        if (PACKED) loadBfrag(Be[0], b2base);
        else {
            const float* bp = W2e + (size_t)(ch * CHUNK + kup * 16) * NC + wv * 96 + lh;
#pragma unroll
            for (int k = 0; k < 16; ++k) Be[0].b[k] = (__bf16)bp[(size_t)k * NC];
        }
#pragma unroll
        for (int idx = 0; idx < 24; ++idx) {
            const int ks = idx / 6, t = idx % 6;
            const int cur = idx & 1;
            if (idx + 1 < 24) {                      // prefetch next B
                const int ks2 = (idx + 1) / 6, t2 = (idx + 1) % 6;
                if (PACKED)
                    loadBfrag(Be[cur ^ 1],
                              b2base + ((size_t)ks2 * (NC / 16) + t2) * 512);
                else {
                    const float* bp = W2e +
                        (size_t)(ch * CHUNK + ks2 * 32 + kup * 16) * NC + (wv * 6 + t2) * 16 + lh;
#pragma unroll
                    for (int k = 0; k < 16; ++k) Be[cur ^ 1].b[k] = (__bf16)bp[(size_t)k * NC];
                }
            }
            if (t == 0) {                            // new K tile: reload A pair
#pragma unroll
                for (int r = 0; r < 2; ++r)
                    loadAfrag(A2[r], &sh[(r * 16 + lh) * LDH + ks * 32 + kup * 16]);
            }
#pragma unroll
            for (int r = 0; r < 2; ++r)
                yacc[r][t] = __builtin_amdgcn_wmma_f32_16x16x32_bf16(
                    false, A2[r].v, false, Be[cur].v, (short)0, yacc[r][t], false, false);
        }
        __syncthreads();
    }

    // ---- epilogue: (y + b2) * top_prob, scatter to token rows ----
#pragma unroll
    for (int r = 0; r < 2; ++r) {
        float pr[8]; int tk[8];
#pragma unroll
        for (int v = 0; v < 8; ++v) {
            int m = r * 16 + kup * 8 + v;
            pr[v] = sprob[m];
            tk[v] = stok[m];
        }
#pragma unroll
        for (int t = 0; t < 6; ++t) {
            const int n2 = (wv * 6 + t) * 16 + lh;
            const float bias2 = b2e[n2];
#pragma unroll
            for (int v = 0; v < 8; ++v) {
                int m = r * 16 + kup * 8 + v;
                if (m < nTok)
                    out[(size_t)tk[v] * NC + n2] = (yacc[r][t][v] + bias2) * pr[v];
            }
        }
    }
}

extern "C" void kernel_launch(void* const* d_in, const int* in_sizes, int n_in,
                              void* d_out, int out_size, void* d_ws, size_t ws_size,
                              hipStream_t stream) {
    const float* x  = (const float*)d_in[0];
    const float* Wr = (const float*)d_in[1];
    const float* W1 = (const float*)d_in[2];
    const float* b1 = (const float*)d_in[3];
    const float* W2 = (const float*)d_in[4];
    const float* b2 = (const float*)d_in[5];

    float* y       = (float*)d_out;
    float* aux_out = y + (size_t)NT * NC;

    char* w = (char*)d_ws;
    int*    cnts    = (int*)(w);                      // 8 ints
    float*  impPart = (float*)(w + 256);              // (NT/8)*8 floats = 32KB
    float*  tprob   = (float*)(w + 256 + 32768);      // NT floats = 32KB
    int*    elist   = (int*)(w + 256 + 65536);        // NE*NT ints = 256KB
    __bf16* w1p     = (__bf16*)(w + 327936);          // 37,748,736 B
    __bf16* w2p     = (__bf16*)(w + 327936 + 37748736);
    const size_t REQ = 327936 + 2ull * 37748736;      // ~75.8 MB
    (void)in_sizes; (void)n_in; (void)out_size;

    moe_init_kernel<<<1, 32, 0, stream>>>(cnts);
    moe_router_kernel<<<NT / 8, 256, 0, stream>>>(x, Wr, cnts, elist, tprob, impPart);
    moe_aux_kernel<<<1, 32, 0, stream>>>(impPart, cnts, aux_out);

    if (ws_size >= REQ) {
        // pack W1: ktiles=24, ntiles=192; W2: ktiles=96, ntiles=48
        const int lanesW = NE * (NC / 32) * (NH / 16) * 32;   // 1,179,648 (same for W2)
        moe_pack_kernel<<<lanesW / 256, 256, 0, stream>>>(W1, w1p, NC / 32, NH / 16, NH);
        moe_pack_kernel<<<lanesW / 256, 256, 0, stream>>>(W2, w2p, NH / 32, NC / 16, NC);
        moe_ffn_kernel<true><<<NE * 256, 256, 0, stream>>>(
            x, W1, b1, W2, b2, w1p, w2p, cnts, elist, tprob, y);
    } else {
        moe_ffn_kernel<false><<<NE * 256, 256, 0, stream>>>(
            x, W1, b1, W2, b2, w1p, w2p, cnts, elist, tprob, y);
    }
}
